// DySAT_8899172237850
// MI455X (gfx1250) — matrix operand order repaired and checked by reading, not compiled, not run on hardware
//
#include <hip/hip_runtime.h>
#include <hip/hip_bf16.h>
#include <math.h>

// ---------------------------------------------------------------------------
// DySAT forward for MI455X (gfx1250, wave32, WMMA + TDM).
//
//   h1pre = x @ W1                 (WMMA bf16, B-tile staged in LDS via TDM)
//   GAT1 edge softmax/aggregate    (2 edge passes, f32 atomics, L2-resident h)
//   h1   = ELU(agg/den + b1) -> bf16
//   h2pre= h1 @ W2
//   GAT2 edge softmax/aggregate, head-mean + b2 + pe[4] -> q
//   ao   = (q@wv+bv)@wo+bo         (temporal attn mask => one-hot => V[:,L-1])
//   y1=LN(q+ao); t=relu(y1@f1+fb1); y2=LN(y1+t@f2+fb2); out=y2@wc+bc
// ---------------------------------------------------------------------------

#define NN    16384
#define DINN  128
#define EE    262144
#define ETOT  (2 * EE + NN)   // symmetrized edges + self loops = 540672
#define HDIM  512             // 4 heads * 128
#define DD    128

typedef __attribute__((ext_vector_type(16))) __bf16 v16bf;
typedef __attribute__((ext_vector_type(8)))  __bf16 v8bf;
typedef __attribute__((ext_vector_type(8)))  float  v8f;
typedef unsigned int u32x4 __attribute__((ext_vector_type(4)));
typedef int          i32x4 __attribute__((ext_vector_type(4)));
typedef int          i32x8 __attribute__((ext_vector_type(8)));

// LDS B-tile staging: TDM pads 32B after every row's payload (pad_interval ==
// one row), giving a simple padded row-major layout with stride KH+16 elems.
// Affine addressing -> ds_load immediate offsets; rows rotate 8 LDS banks.
#define LDS_BUF_ELEMS 17408   // 64 rows * (256+16) elems = 34816B per buffer

// ---- Tensor Data Mover: 2-D tile (64 rows x KH elems) of WT into LDS ------
// D# per CDNA5 ISA ch.8: group0 = flags/lds/global addr, group1 = dims/strides.
__device__ __forceinline__ void tdm_load_tile(unsigned lds_byte_addr,
                                              const __bf16* gsrc,
                                              int KH, int K) {
  unsigned long long ga = (unsigned long long)(uintptr_t)gsrc;
  u32x4 g0;
  g0[0] = 1u;                                   // count=1 (valid), user mode
  g0[1] = lds_byte_addr;                        // LDS dest (bytes)
  g0[2] = (unsigned)(ga & 0xffffffffu);         // global_addr[31:0]
  g0[3] = (unsigned)(ga >> 32) | (2u << 30);    // global_addr[56:32] | type=2
  // pad_interval encodes 2^(v+1) dwords; one row = KH/2 dwords -> v = log2(KH/2)-1
  unsigned ivl = (KH == 256) ? 6u : 4u;         // 512B row vs 128B row
  i32x8 g1;
  g1[0] = (int)((1u << 16) |                    // data_size = 2 bytes
                (1u << 20) |                    // pad_enable
                (ivl << 22) |                   // pad_interval = one row payload
                (7u << 25));                    // pad_amount  = 8 dw = 32B
  g1[1] = (int)((unsigned)(KH & 0xffff) << 16); // tensor_dim0[15:0]
  g1[2] = (int)(64u << 16);                     // dim0[31:16]=0, tensor_dim1=64
  g1[3] = (int)((unsigned)(KH & 0xffff) << 16); // dim1[31:16]=0, tile_dim0=KH
  g1[4] = 64;                                   // tile_dim1 = 64, tile_dim2 = 0
  g1[5] = K;                                    // tensor_dim0_stride[31:0] = K
  g1[6] = 0;                                    // stride0 hi, dim1_stride lo
  g1[7] = 0;
  i32x4 gz = {0, 0, 0, 0};
#if defined(__clang_major__) && (__clang_major__ >= 23)
  i32x8 gz8 = {0, 0, 0, 0, 0, 0, 0, 0};
  __builtin_amdgcn_tensor_load_to_lds(g0, g1, gz, gz, gz8, 0);
#else
  __builtin_amdgcn_tensor_load_to_lds(g0, g1, gz, gz, 0);
#endif
}

// ---------------- order-preserving float<->uint encode (for atomicMax) -----
__device__ __forceinline__ unsigned encf(float f) {
  unsigned u = __float_as_uint(f);
  return (u & 0x80000000u) ? ~u : (u | 0x80000000u);
}
__device__ __forceinline__ float decf(unsigned u) {
  return __uint_as_float((u & 0x80000000u) ? (u & 0x7fffffffu) : ~u);
}

__device__ __forceinline__ void get_edge(const int* __restrict__ ei, int i,
                                         int& s, int& d) {
  if (i < EE)          { s = ei[i];             d = ei[EE + i]; }
  else if (i < 2 * EE) { s = ei[EE + (i - EE)]; d = ei[i - EE]; }
  else                 { s = i - 2 * EE;        d = s; }
}

// ---------------- elementwise convert / transpose --------------------------
__global__ __launch_bounds__(256) void tobf16_k(const float* __restrict__ x,
                                                __bf16* __restrict__ y, int n) {
  int i = blockIdx.x * 256 + threadIdx.x;
  if (i < n) y[i] = (__bf16)x[i];
}

// W: K x Ncols row-major  ->  WT: Ncols x K row-major (bf16)
__global__ __launch_bounds__(256) void transpose_bf16_k(const float* __restrict__ W,
                                                        __bf16* __restrict__ WT,
                                                        int K, int Ncols) {
  int i = blockIdx.x * 256 + threadIdx.x;
  if (i >= K * Ncols) return;
  int k = i / Ncols, c = i % Ncols;
  WT[(size_t)c * K + k] = (__bf16)W[(size_t)k * Ncols + c];
}

// ---------------- WMMA bf16 GEMM with TDM-staged B tile --------------------
// C[NN x Ncols] = A[NN x K] @ WT^T ; WT is [Ncols x K] bf16.
// Block = 256 thr = 8 waves; wave w -> rows [by*128+16w,+16), cols [bx*64,+64).
// B tile (64 x K) is DMA'd into LDS by the TDM in two K-halves, double
// buffered: half-1 DMA overlaps half-0 WMMA compute.
// Fragment layouts used here (contiguous k, A and B alike):
//   A lane(l): m=l&15, 16 contiguous K elems starting at (l>>4)*16
//   B lane(l): n=l&15, 16 contiguous K elems starting at (l>>4)*16
//   C vgpr v : m = v + 8*(l>>4), n = l&15
__global__ __launch_bounds__(256) void wmma_gemm_k(const __bf16* __restrict__ A,
                                                   const __bf16* __restrict__ WT,
                                                   const float*  __restrict__ bias,
                                                   float*  __restrict__ Cf,
                                                   __bf16* __restrict__ Cbf,
                                                   int K, int Ncols, int act) {
  __shared__ __align__(64) __bf16 ldsB[2 * LDS_BUF_ELEMS];
  const int wave = threadIdx.x >> 5;
  const int lane = threadIdx.x & 31;
  const int lm = lane & 15;
  const int lh = lane >> 4;
  const int r0 = blockIdx.y * 128 + wave * 16;
  const int colBase = blockIdx.x * 64;
  const int KH = K >> 1;
  const int RS = KH + 16;                 // padded LDS row stride (elements)

  const __bf16* arow = A + (size_t)(r0 + lm) * K;
  const __bf16* gtile = WT + (size_t)colBase * K;
  const unsigned ldsbase = (unsigned)(uintptr_t)(void*)&ldsB[0];

  // TDM prologue: fetch half 0, then kick half 1 asynchronously.
  if (wave == 0) {
    tdm_load_tile(ldsbase, gtile, KH, K);
    __builtin_amdgcn_s_wait_tensorcnt(0);
  }
  __syncthreads();
  if (wave == 0) {
    tdm_load_tile(ldsbase + 2u * LDS_BUF_ELEMS, gtile + KH, KH, K);
  }

  v8f c0 = {}, c1 = {}, c2 = {}, c3 = {};
#pragma unroll 1
  for (int h = 0; h < 2; ++h) {
    if (h == 1) {
      if (wave == 0) __builtin_amdgcn_s_wait_tensorcnt(0);
      __syncthreads();
    }
    // Per-row base pointers (affine in k0 -> immediate ds offsets).
    const __bf16* bp = &ldsB[h * LDS_BUF_ELEMS] + lm * RS + lh * 16;
    const int rs16 = 16 * RS;
    const __bf16* ar = arow + h * KH;
    for (int k0 = 0; k0 < KH; k0 += 32) {
      const int kb = k0 + lh * 16;
      v8bf alo = *(const v8bf*)(ar + kb);
      v8bf ahi = *(const v8bf*)(ar + kb + 8);
      v16bf a = __builtin_shufflevector(alo, ahi, 0, 1, 2, 3, 4, 5, 6, 7,
                                        8, 9, 10, 11, 12, 13, 14, 15);
      v16bf b0 = *(const v16bf*)(bp + k0);
      v16bf b1 = *(const v16bf*)(bp + rs16 + k0);
      v16bf b2 = *(const v16bf*)(bp + 2 * rs16 + k0);
      v16bf b3 = *(const v16bf*)(bp + 3 * rs16 + k0);
      c0 = __builtin_amdgcn_wmma_f32_16x16x32_bf16(false, a, false, b0, (short)0, c0, false, false);
      c1 = __builtin_amdgcn_wmma_f32_16x16x32_bf16(false, a, false, b1, (short)0, c1, false, false);
      c2 = __builtin_amdgcn_wmma_f32_16x16x32_bf16(false, a, false, b2, (short)0, c2, false, false);
      c3 = __builtin_amdgcn_wmma_f32_16x16x32_bf16(false, a, false, b3, (short)0, c3, false, false);
    }
  }

  v8f cc[4] = {c0, c1, c2, c3};
#pragma unroll
  for (int cb = 0; cb < 4; ++cb) {
    const int col = colBase + cb * 16 + lm;
    const float bvv = bias ? bias[col] : 0.0f;
#pragma unroll
    for (int v = 0; v < 8; ++v) {
      float val = cc[cb][v] + bvv;
      if (act == 1) val = fmaxf(val, 0.0f);
      const size_t o = (size_t)(r0 + v + lh * 8) * Ncols + col;
      if (Cf)  Cf[o]  = val;
      if (Cbf) Cbf[o] = (__bf16)val;
    }
  }
}

// ---------------- GAT: per-node attention logits (wave per (n,head)) -------
__global__ __launch_bounds__(256) void attn_logits_k(const float* __restrict__ h,
                                                     const float* __restrict__ a_src,
                                                     const float* __restrict__ a_dst,
                                                     float* __restrict__ als,
                                                     float* __restrict__ ald) {
  int t = blockIdx.x * 8 + (threadIdx.x >> 5);
  int lane = threadIdx.x & 31;
  if (t >= NN * 4) return;
  int n = t >> 2, hh = t & 3;
  const float* hp = h + (size_t)n * HDIM + hh * 128;
  const float* as = a_src + hh * 128;
  const float* ad = a_dst + hh * 128;
  float s1 = 0.f, s2 = 0.f;
#pragma unroll
  for (int j = 0; j < 4; ++j) {
    int d = lane + j * 32;
    float v = hp[d];
    s1 += v * as[d];
    s2 += v * ad[d];
  }
  for (int o = 16; o > 0; o >>= 1) {
    s1 += __shfl_xor(s1, o, 32);
    s2 += __shfl_xor(s2, o, 32);
  }
  if (lane == 0) { als[t] = s1; ald[t] = s2; }
}

// ---------------- GAT: edge pass 1 (segment max of leaky_relu logits) ------
__global__ __launch_bounds__(256) void edge_max_k(const int* __restrict__ ei,
                                                  const float* __restrict__ als,
                                                  const float* __restrict__ ald,
                                                  unsigned* __restrict__ mEnc) {
  int i = blockIdx.x * 256 + threadIdx.x;
  if (i >= ETOT) return;
  int s, d; get_edge(ei, i, s, d);
#pragma unroll
  for (int h = 0; h < 4; ++h) {
    float e = als[s * 4 + h] + ald[d * 4 + h];
    e = (e >= 0.f) ? e : 0.2f * e;
    atomicMax(&mEnc[d * 4 + h], encf(e));
  }
}

// ---------------- GAT: edge pass 2 (sum exp, sum exp*h[src]) ---------------
__global__ __launch_bounds__(256) void edge_acc_k(const int* __restrict__ ei,
                                                  const float* __restrict__ als,
                                                  const float* __restrict__ ald,
                                                  const unsigned* __restrict__ mEnc,
                                                  const float* __restrict__ h,
                                                  float* __restrict__ acc,
                                                  float* __restrict__ den) {
  int gid = blockIdx.x * 256 + threadIdx.x;
  int edge = gid >> 5, lane = gid & 31;
  if (edge >= ETOT) return;
  int s, d; get_edge(ei, edge, s, d);
  const float* hs = h + (size_t)s * HDIM;
  __builtin_prefetch(hs, 0, 3);   // global_prefetch_b8
  float ex[4];
#pragma unroll
  for (int hh = 0; hh < 4; ++hh) {
    float e = als[s * 4 + hh] + ald[d * 4 + hh];
    e = (e >= 0.f) ? e : 0.2f * e;
    ex[hh] = __expf(e - decf(mEnc[d * 4 + hh]));
  }
  if (lane < 4) atomicAdd(&den[d * 4 + lane], ex[lane]);
  float* ad = acc + (size_t)d * HDIM;
#pragma unroll
  for (int j = 0; j < 16; ++j) {
    int idx = lane + j * 32;
    atomicAdd(&ad[idx], ex[idx >> 7] * hs[idx]);
  }
}

// ---------------- layer-1 finalize: normalize + b1 + ELU -> bf16 -----------
__global__ __launch_bounds__(256) void finalize1_k(const float* __restrict__ acc,
                                                   const float* __restrict__ den,
                                                   const float* __restrict__ b1,
                                                   __bf16* __restrict__ hbf) {
  int i = blockIdx.x * 256 + threadIdx.x;
  if (i >= NN * HDIM) return;
  int n = i >> 9, j = i & 511, h = j >> 7;
  float v = acc[i] / (den[n * 4 + h] + 1e-16f) + b1[j];
  v = (v > 0.f) ? v : (__expf(v) - 1.f);  // ELU
  hbf[i] = (__bf16)v;
}

// ---------------- layer-2 finalize: head-mean + b2 + pe[L-1] -> q ----------
__global__ __launch_bounds__(256) void finalize2_k(const float* __restrict__ acc,
                                                   const float* __restrict__ den,
                                                   const float* __restrict__ b2,
                                                   float* __restrict__ q,
                                                   __bf16* __restrict__ qbf) {
  int i = blockIdx.x * 256 + threadIdx.x;
  if (i >= NN * DD) return;
  int n = i >> 7, d = i & 127;
  float s = 0.f;
#pragma unroll
  for (int h = 0; h < 4; ++h)
    s += acc[(size_t)n * HDIM + h * 128 + d] / (den[n * 4 + h] + 1e-16f);
  s = 0.25f * s + b2[d];
  int pair = d & ~1;
  float div = __expf((float)pair * (-logf(10000.f) / 128.f));
  s += (d & 1) ? __cosf(4.f * div) : __sinf(4.f * div);
  q[i] = s;
  qbf[i] = (__bf16)s;
}

// ---------------- LayerNorm( r1 + r2 ) * g + b  (one wave per node) --------
__global__ __launch_bounds__(256) void layernorm_k(const float* __restrict__ r1,
                                                   const float* __restrict__ r2,
                                                   const float* __restrict__ g,
                                                   const float* __restrict__ b,
                                                   float* __restrict__ y,
                                                   __bf16* __restrict__ ybf) {
  int n = blockIdx.x * 8 + (threadIdx.x >> 5);
  int lane = threadIdx.x & 31;
  if (n >= NN) return;
  float v[4]; float sum = 0.f;
#pragma unroll
  for (int j = 0; j < 4; ++j) {
    int d = lane + j * 32;
    v[j] = r1[(size_t)n * DD + d] + r2[(size_t)n * DD + d];
    sum += v[j];
  }
  for (int o = 16; o > 0; o >>= 1) sum += __shfl_xor(sum, o, 32);
  float mu = sum * (1.f / 128.f);
  float s2 = 0.f;
#pragma unroll
  for (int j = 0; j < 4; ++j) { float t = v[j] - mu; s2 += t * t; }
  for (int o = 16; o > 0; o >>= 1) s2 += __shfl_xor(s2, o, 32);
  float inv = rsqrtf(s2 * (1.f / 128.f) + 1e-5f);
#pragma unroll
  for (int j = 0; j < 4; ++j) {
    int d = lane + j * 32;
    float o = (v[j] - mu) * inv * g[d] + b[d];
    y[(size_t)n * DD + d] = o;
    if (ybf) ybf[(size_t)n * DD + d] = (__bf16)o;
  }
}

// ---------------- out = y2 @ wc + bc  (OUT=2) ------------------------------
__global__ __launch_bounds__(256) void outproj_k(const float* __restrict__ y2,
                                                 const float* __restrict__ wc,
                                                 const float* __restrict__ bc,
                                                 float* __restrict__ out) {
  int n = blockIdx.x * 256 + threadIdx.x;
  if (n >= NN) return;
  float a0 = bc[0], a1 = bc[1];
  const float* yr = y2 + (size_t)n * DD;
#pragma unroll 4
  for (int d = 0; d < 128; ++d) {
    float v = yr[d];
    a0 += v * wc[d * 2 + 0];
    a1 += v * wc[d * 2 + 1];
  }
  out[n * 2 + 0] = a0;
  out[n * 2 + 1] = a1;
}

// ===========================================================================
extern "C" void kernel_launch(void* const* d_in, const int* in_sizes, int n_in,
                              void* d_out, int out_size, void* d_ws, size_t ws_size,
                              hipStream_t stream) {
  (void)in_sizes; (void)n_in; (void)out_size; (void)ws_size;
  const float* x      = (const float*)d_in[0];
  const int*   ei     = (const int*)d_in[1];
  const float* W1     = (const float*)d_in[3];
  const float* a_src1 = (const float*)d_in[4];
  const float* a_dst1 = (const float*)d_in[5];
  const float* b1     = (const float*)d_in[6];
  const float* W2     = (const float*)d_in[7];
  const float* a_src2 = (const float*)d_in[8];
  const float* a_dst2 = (const float*)d_in[9];
  const float* b2     = (const float*)d_in[10];
  const float* wv     = (const float*)d_in[15];
  const float* bvp    = (const float*)d_in[16];
  const float* wo     = (const float*)d_in[17];
  const float* bo     = (const float*)d_in[18];
  const float* ln_g   = (const float*)d_in[19];
  const float* ln_b   = (const float*)d_in[20];
  const float* f1     = (const float*)d_in[21];
  const float* fb1    = (const float*)d_in[22];
  const float* f2     = (const float*)d_in[23];
  const float* fb2    = (const float*)d_in[24];
  const float* wc     = (const float*)d_in[25];
  const float* bc     = (const float*)d_in[26];

  char* ws = (char*)d_ws;
  const size_t MB = 1024 * 1024;
  // Region map (with lifetime-based reuse):
  float*  hpre = (float*)(ws + 0);            // 32MB: h1pre / h2pre
  float*  acc  = (float*)(ws + 32 * MB);      // 32MB: edge accumulators
  __bf16* hbf  = (__bf16*)(ws + 64 * MB);     // 16MB: h1 bf16 / later tbf
  __bf16* xbf  = (__bf16*)(ws + 80 * MB);     //  4MB: x bf16 / later qbf
  float*  q    = (float*)(ws + 84 * MB);      //  8MB
  float*  y1   = (float*)(ws + 92 * MB);      //  8MB
  // reuse aliases (lifetimes disjoint with originals):
  float*  ao   = (float*)(ws + 0 + 8 * MB);
  float*  u    = (float*)(ws + 0 + 16 * MB);
  float*  y2   = (float*)(ws + 0 + 24 * MB);
  __bf16* vbf  = (__bf16*)(ws + 32 * MB);
  __bf16* y1bf = (__bf16*)(ws + 32 * MB + 4 * MB);
  __bf16* tbf  = (__bf16*)(ws + 64 * MB);
  __bf16* qbf  = (__bf16*)(ws + 80 * MB);
  // small buffers
  size_t o = 100 * MB;
  __bf16* W2T = (__bf16*)(ws + o); o += 524288;
  __bf16* W1T = (__bf16*)(ws + o); o += 131072;
  __bf16* F1T = (__bf16*)(ws + o); o += 131072;
  __bf16* F2T = (__bf16*)(ws + o); o += 131072;
  __bf16* WVT = (__bf16*)(ws + o); o += 32768;
  __bf16* WOT = (__bf16*)(ws + o); o += 32768;
  float* als  = (float*)(ws + o); o += 262144;
  float* ald  = (float*)(ws + o); o += 262144;
  unsigned* mEnc = (unsigned*)(ws + o); o += 262144;
  float* den  = (float*)(ws + o); o += 262144;

  auto cd = [](int n) { return (n + 255) / 256; };

  // ---- prep: bf16 convert + weight transposes -----------------------------
  tobf16_k<<<cd(NN * DINN), 256, 0, stream>>>(x, xbf, NN * DINN);
  transpose_bf16_k<<<cd(128 * 512), 256, 0, stream>>>(W1, W1T, 128, 512);
  transpose_bf16_k<<<cd(512 * 512), 256, 0, stream>>>(W2, W2T, 512, 512);
  transpose_bf16_k<<<cd(128 * 128), 256, 0, stream>>>(wv, WVT, 128, 128);
  transpose_bf16_k<<<cd(128 * 128), 256, 0, stream>>>(wo, WOT, 128, 128);
  transpose_bf16_k<<<cd(128 * 512), 256, 0, stream>>>(f1, F1T, 128, 512);
  transpose_bf16_k<<<cd(512 * 128), 256, 0, stream>>>(f2, F2T, 512, 128);

  // ---- GAT layer 1 --------------------------------------------------------
  wmma_gemm_k<<<dim3(8, 128), 256, 0, stream>>>(xbf, W1T, nullptr, hpre, nullptr, 128, 512, 0);
  attn_logits_k<<<NN * 4 / 8, 256, 0, stream>>>(hpre, a_src1, a_dst1, als, ald);
  hipMemsetAsync(mEnc, 0, NN * 4 * 4, stream);
  hipMemsetAsync(den, 0, NN * 4 * 4, stream);
  hipMemsetAsync(acc, 0, (size_t)NN * HDIM * 4, stream);
  edge_max_k<<<cd(ETOT), 256, 0, stream>>>(ei, als, ald, mEnc);
  edge_acc_k<<<ETOT / 8, 256, 0, stream>>>(ei, als, ald, mEnc, hpre, acc, den);
  finalize1_k<<<cd(NN * HDIM), 256, 0, stream>>>(acc, den, b1, hbf);

  // ---- GAT layer 2 --------------------------------------------------------
  wmma_gemm_k<<<dim3(8, 128), 256, 0, stream>>>(hbf, W2T, nullptr, hpre, nullptr, 512, 512, 0);
  attn_logits_k<<<NN * 4 / 8, 256, 0, stream>>>(hpre, a_src2, a_dst2, als, ald);
  hipMemsetAsync(mEnc, 0, NN * 4 * 4, stream);
  hipMemsetAsync(den, 0, NN * 4 * 4, stream);
  hipMemsetAsync(acc, 0, (size_t)NN * HDIM * 4, stream);
  edge_max_k<<<cd(ETOT), 256, 0, stream>>>(ei, als, ald, mEnc);
  edge_acc_k<<<ETOT / 8, 256, 0, stream>>>(ei, als, ald, mEnc, hpre, acc, den);
  finalize2_k<<<cd(NN * DD), 256, 0, stream>>>(acc, den, b2, q, qbf);

  // ---- temporal attn (one-hot) + output head ------------------------------
  wmma_gemm_k<<<dim3(2, 128), 256, 0, stream>>>(qbf, WVT, bvp, nullptr, vbf, 128, 128, 0);
  wmma_gemm_k<<<dim3(2, 128), 256, 0, stream>>>(vbf, WOT, bo, ao, nullptr, 128, 128, 0);
  layernorm_k<<<NN / 8, 256, 0, stream>>>(q, ao, ln_g, ln_b, y1, y1bf);
  wmma_gemm_k<<<dim3(8, 128), 256, 0, stream>>>(y1bf, F1T, fb1, nullptr, tbf, 128, 512, 1);
  wmma_gemm_k<<<dim3(2, 128), 256, 0, stream>>>(tbf, F2T, fb2, u, nullptr, 512, 128, 0);
  layernorm_k<<<NN / 8, 256, 0, stream>>>(y1, u, ln_g, ln_b, y2, nullptr);
  outproj_k<<<cd(NN), 256, 0, stream>>>(y2, wc, bc, (float*)d_out);
}
